// FrameSimilarity_33045478376105
// MI455X (gfx1250) — compile-verified, matrix operand order checked by reading, not run on hardware
//
#include <hip/hip_runtime.h>
#include <hip/hip_bf16.h>

// ---------------------------------------------------------------------------
// FrameSimilarity on MI455X (gfx1250, wave32, WMMA)
//
// Roofline: dominant cost is streaming x1/x2 (268 MB fp32) -> ~11.5 us at
// 23.3 TB/s. All three GEMMs total < 1.5 GFLOP -> negligible on WMMA units.
// Strategy:
//   k_prep      : W_proj -> W_projT (f16), W_fc -> W_fcT (f16, K padded 101->128)
//   k_pool_proj : pool(x1,x2) via NON-TEMPORAL b128 loads (one-pass 268 MB
//                 stream > 192 MB L2 -> don't pollute L2) -> LDS f16 (16x512)
//                 -> WMMA proj (K=512) -> L2-normalize -> Y f16 (B,T,128)
//                 [2 MB, stays hot in L2 for kernel 2's ~8x reuse]
//   k_band_fc   : banded gram via 8 WMMA tiles of Y_t x Y_s^T (K=128),
//                 scatter to 16x128 LDS band, WMMA band@W_fcT (K=128),
//                 +bias, ReLU -> out f32 (non-temporal stores, write-once)
// ---------------------------------------------------------------------------

typedef __attribute__((ext_vector_type(16))) _Float16 v16h;
typedef __attribute__((ext_vector_type(8)))  _Float16 v8h;
typedef __attribute__((ext_vector_type(8)))  float    v8f;
typedef __attribute__((ext_vector_type(4)))  float    v4f;

static constexpr int Bsz = 4;
static constexpr int Tsz = 2048;
static constexpr int Csz = 256;     // channels per tensor
static constexpr int KPROJ = 512;   // 2*C
static constexpr int DIM = 128;     // SIM_DIM == OUT_DIM
static constexpr int HALF = 50;     // (101-1)/2

// A/B fragment layout for v_wmma_f32_16x16x32_f16 (ISA 7.12.2):
// lane L: row/col = L&15, hi = L>>4.
//   elems 0..7  <- K = kb + hi*8 + j
//   elems 8..15 <- K = kb + 16 + hi*8 + j
__device__ __forceinline__ v16h load_frag(const _Float16* p) {
  v8h lo = *(const v8h*)(p);
  v8h hi = *(const v8h*)(p + 16);
  v16h r;
#pragma unroll
  for (int i = 0; i < 8; ++i) { r[i] = lo[i]; r[i + 8] = hi[i]; }
  return r;
}

__device__ __forceinline__ v8f wmma16(v16h a, v16h b, v8f c) {
  return __builtin_amdgcn_wmma_f32_16x16x32_f16(
      /*neg_a=*/false, a, /*neg_b=*/false, b,
      /*c_mod=*/(short)0, c, /*reuse_a=*/false, /*reuse_b=*/false);
}

// ---------------------------------------------------------------------------
// Kernel 0: transpose + fp16-convert the weight matrices (tiny, L2 resident).
// W_projT[n][k] = W_proj[k][n]   (128 x 512)
// W_fcT  [n][k] = k<101 ? W_fc[k][n] : 0   (128 x 128, K zero-padded)
// ---------------------------------------------------------------------------
__global__ void k_prep(const float* __restrict__ Wproj,
                       const float* __restrict__ Wfc,
                       _Float16* __restrict__ WprojT,
                       _Float16* __restrict__ WfcT) {
  int idx = blockIdx.x * 256 + threadIdx.x;
  if (idx < DIM * KPROJ) {
    int n = idx >> 9, k = idx & (KPROJ - 1);
    WprojT[idx] = (_Float16)Wproj[k * DIM + n];
  }
  int idx2 = idx - DIM * KPROJ;
  if (idx2 >= 0 && idx2 < DIM * DIM) {
    int n = idx2 >> 7, k = idx2 & (DIM - 1);
    WfcT[idx2] = (_Float16)((k < 101) ? Wfc[k * DIM + n] : 0.f);
  }
}

// ---------------------------------------------------------------------------
// Kernel 1: pool + project + normalize. One block = 16 consecutive t rows.
// 256 threads = 8 waves; wave j owns output N-tile [16j, 16j+16).
// ---------------------------------------------------------------------------
__global__ void __launch_bounds__(256)
k_pool_proj(const float* __restrict__ x1, const float* __restrict__ x2,
            const _Float16* __restrict__ WprojT, _Float16* __restrict__ Y) {
  __shared__ _Float16 pooled[16][KPROJ];   // 16 KB
  __shared__ float    ytile[16][DIM];      // 8 KB
  __shared__ float    rnorm[16];

  const int tile = blockIdx.x;
  const int b  = tile >> 7;                // tile / (T/16)
  const int t0 = (tile & 127) << 4;
  const int tid = threadIdx.x;

  // --- pooling: non-temporal float4 coalesced streaming loads --------------
  // thread -> quarter (rows 4q..4q+3), channel group c4 = (tid&63)*4
  const int q  = tid >> 6;
  const int c4 = (tid & 63) << 2;
  const size_t rowbase = (size_t)(b * Tsz + t0) * 16 * Csz;
#pragma unroll
  for (int rr = 0; rr < 4; ++rr) {
    const int r = q * 4 + rr;
    const v4f* p1 = (const v4f*)(x1 + rowbase + (size_t)r * 16 * Csz + c4);
    const v4f* p2 = (const v4f*)(x2 + rowbase + (size_t)r * 16 * Csz + c4);
    v4f s1 = {};
    v4f s2 = {};
#pragma unroll
    for (int hw = 0; hw < 16; ++hw) {
      v4f v1 = __builtin_nontemporal_load(p1 + hw * (Csz / 4));
      v4f v2 = __builtin_nontemporal_load(p2 + hw * (Csz / 4));
      s1 += v1;
      s2 += v2;
    }
#pragma unroll
    for (int i = 0; i < 4; ++i) {
      pooled[r][c4 + i]       = (_Float16)(s1[i] * 0.0625f);
      pooled[r][Csz + c4 + i] = (_Float16)(s2[i] * 0.0625f);
    }
  }
  __syncthreads();

  // --- projection: D(16x16) += A(16x32) x B(32x16), K = 512 ----------------
  const int wave = tid >> 5;
  const int lane = tid & 31;
  const int rw   = lane & 15;
  const int hi   = lane >> 4;
  v8f acc = {};
#pragma unroll 4
  for (int kb = 0; kb < KPROJ; kb += 32) {
    v16h a = load_frag(&pooled[rw][kb + hi * 8]);
    v16h bf = load_frag(WprojT + (size_t)(16 * wave + rw) * KPROJ + kb + hi * 8);
    acc = wmma16(a, bf, acc);
  }
  // C layout: VGPR v, lanes 0-15 -> M=v, lanes 16-31 -> M=v+8; N = lane&15
#pragma unroll
  for (int v = 0; v < 8; ++v) ytile[v + 8 * hi][16 * wave + rw] = acc[v];
  __syncthreads();

  if (tid < 16) {
    float s = 0.f;
#pragma unroll 8
    for (int i = 0; i < DIM; ++i) { float x = ytile[tid][i]; s += x * x; }
    rnorm[tid] = rsqrtf(s + 1e-12f);
  }
  __syncthreads();

  _Float16* yo = Y + (size_t)(b * Tsz + t0) * DIM;
#pragma unroll
  for (int i = tid; i < 16 * DIM; i += 256) {
    int r = i >> 7;
    yo[i] = (_Float16)(ytile[r][i & (DIM - 1)] * rnorm[r]);
  }
}

// ---------------------------------------------------------------------------
// Kernel 2: banded gram (A x B^T via row loads of row-major Y) + band@W_fc.
// Wave j computes sim tile: rows [t0,t0+16) x cols s in [t0-50+16j, +16).
// band[m][w] with w = 16j + n - m (unique per (m,w)); K zero-padded to 128.
// ---------------------------------------------------------------------------
__global__ void __launch_bounds__(256)
k_band_fc(const _Float16* __restrict__ Y, const _Float16* __restrict__ WfcT,
          const float* __restrict__ bfc, float* __restrict__ out) {
  __shared__ _Float16 band[16][DIM];       // 4 KB, zero-padded 101 -> 128

  const int tile = blockIdx.x;
  const int b  = tile >> 7;
  const int t0 = (tile & 127) << 4;
  const int tid  = threadIdx.x;
  const int wave = tid >> 5;
  const int lane = tid & 31;
  const int rw   = lane & 15;
  const int hi   = lane >> 4;

  _Float16* bflat = &band[0][0];
#pragma unroll
  for (int i = tid; i < 16 * DIM; i += 256) bflat[i] = (_Float16)0.f;
  __syncthreads();

  const _Float16* Yb = Y + (size_t)b * Tsz * DIM;
  const int s = t0 - HALF + 16 * wave + rw;       // this lane's gram column
  const bool sv = (s >= 0) && (s < Tsz);          // OOB -> zero row (== pad)

  v8f acc = {};
#pragma unroll
  for (int kb = 0; kb < DIM; kb += 32) {
    v16h a = load_frag(Yb + (size_t)(t0 + rw) * DIM + kb + hi * 8);
    v16h bf = {};
    if (sv) bf = load_frag(Yb + (size_t)s * DIM + kb + hi * 8);
    acc = wmma16(a, bf, acc);
  }
  // scatter sim tile into band: m = v + 8*hi, n = rw, w = 16*wave + n - m
#pragma unroll
  for (int v = 0; v < 8; ++v) {
    int m = v + 8 * hi;
    int w = 16 * wave + rw - m;
    if (w >= 0 && w <= 100) band[m][w] = (_Float16)acc[v];
  }
  __syncthreads();

  // out tile cols [16*wave, +16): D = band(16x128) x W_fc(128x16) via WfcT rows
  v8f acc2 = {};
#pragma unroll
  for (int kb = 0; kb < DIM; kb += 32) {
    v16h a = load_frag(&band[rw][kb + hi * 8]);
    v16h bf = load_frag(WfcT + (size_t)(16 * wave + rw) * DIM + kb + hi * 8);
    acc2 = wmma16(a, bf, acc2);
  }
  const float bias = bfc[16 * wave + rw];
  float* o = out + (size_t)(b * Tsz + t0) * DIM;
#pragma unroll
  for (int v = 0; v < 8; ++v) {
    int m = v + 8 * hi;
    float val = acc2[v] + bias;
    __builtin_nontemporal_store(val > 0.f ? val : 0.f,
                                o + (size_t)m * DIM + 16 * wave + rw);
  }
}

// ---------------------------------------------------------------------------
extern "C" void kernel_launch(void* const* d_in, const int* in_sizes, int n_in,
                              void* d_out, int out_size, void* d_ws, size_t ws_size,
                              hipStream_t stream) {
  const float* x1    = (const float*)d_in[0];
  const float* x2    = (const float*)d_in[1];
  const float* Wproj = (const float*)d_in[2];   // (512,128) f32
  const float* Wfc   = (const float*)d_in[3];   // (101,128) f32
  const float* bfc   = (const float*)d_in[4];   // (128,)    f32
  float* out = (float*)d_out;                   // (4,2048,128) f32

  char* ws = (char*)d_ws;
  _Float16* WprojT = (_Float16*)ws;                                  // 128 KB
  _Float16* WfcT   = (_Float16*)(ws + DIM * KPROJ * sizeof(_Float16)); // 32 KB
  _Float16* Y      = (_Float16*)(ws + (DIM * KPROJ + DIM * DIM) * sizeof(_Float16)); // 2 MB

  const int ntiles = Bsz * Tsz / 16;  // 512
  k_prep<<<(DIM * KPROJ + DIM * DIM + 255) / 256, 256, 0, stream>>>(Wproj, Wfc, WprojT, WfcT);
  k_pool_proj<<<ntiles, 256, 0, stream>>>(x1, x2, WprojT, Y);
  k_band_fc<<<ntiles, 256, 0, stream>>>(Y, WfcT, bfc, out);
}
